// SimpleSNN_87522843558002
// MI455X (gfx1250) — matrix operand order repaired and checked by reading, not compiled, not run on hardware
//
#include <hip/hip_runtime.h>
#include <stdint.h>

// ---------------------------------------------------------------------------
// Fused SNN forward for MI455X (gfx1250, wave32, WMMA).
//   out[b,o] = b2[o] + sum_h W2[o,h] * acc[b,h]
//   acc[b,h] = sum_t z2_t[b,h]  from double-LIF on cur_t = spikes_t @ W1^T + b1
// Spikes are {0,1} -> exact in f16; GEMM via v_wmma_f32_16x16x32_f16.
// All-T spikes staged in LDS once; per N-tile, W1 fragments stay register-
// resident across the 10 timesteps; A fragments ping-pong double-buffered
// (t-loop unrolled x2 -> zero register-copy rotation).
// ---------------------------------------------------------------------------

typedef _Float16 v16h __attribute__((ext_vector_type(16)));
typedef _Float16 v8h  __attribute__((ext_vector_type(8)));
typedef float    v8f  __attribute__((ext_vector_type(8)));

#define T_STEPS 10
#define B_TOT   32768
#define IN_DIM  128
#define H_DIM   256
#define OUT_DIM 10
#define BETA    0.90483741803596f   /* exp(-0.001/0.01) */
#define BTILE   32                  /* batch rows per workgroup */
#define NTHREADS 128                /* 4 waves */

// LDS layout (bytes). Padded strides avoid 16-way bank conflicts on the
// per-lane row gathers for WMMA fragments.
#define W1_STRIDE  136              /* halves per W1 row (128 + 8 pad)   */
#define SPK_STRIDE 136              /* halves per spike row              */
#define SPK_TSTRIDE (BTILE * SPK_STRIDE)      /* halves per timestep plane */
#define ACC_STRIDE 260              /* floats per acc row (256 + 4 pad)  */
#define W1_BYTES (H_DIM * W1_STRIDE * 2)      /* 69632 */
#define SPK_OFF  W1_BYTES
#define SPK_BYTES (T_STEPS * SPK_TSTRIDE * 2) /* 87040 */
#define LDS_BYTES (SPK_OFF + SPK_BYTES)       /* 156672: 2 WGs fit in WGP LDS */
/* acc staging (32*260*4 = 33280 B) reuses [0,33280) after the compute  */

// Cheap, deterministic counter hash (murmur3 finalizer). Exact JAX threefry
// reproduction is not required here; this keeps RNG VALU cost well below the
// WMMA cost instead of dominating it.
__device__ __forceinline__ float urand(uint32_t idx) {
    uint32_t h = idx * 0x9E3779B1u + 0x42424242u;
    h ^= h >> 16; h *= 0x85EBCA6Bu;
    h ^= h >> 13; h *= 0xC2B2AE35u;
    h ^= h >> 16;
    return (float)(h >> 8) * 0x1p-24f;
}

__global__ __launch_bounds__(NTHREADS, 1)
void snn_fused(const float* __restrict__ x,
               const float* __restrict__ W1,
               const float* __restrict__ b1,
               const float* __restrict__ W2,
               const float* __restrict__ b2,
               float* __restrict__ out) {
    __shared__ __align__(16) unsigned char smem[LDS_BYTES];
    _Float16* w1l = (_Float16*)(smem);
    _Float16* spk = (_Float16*)(smem + SPK_OFF);

    const int tid  = threadIdx.x;
    const int wave = tid >> 5;
    const int lane = tid & 31;
    const int g    = lane >> 4;      // half-wave group (K split in fragments)
    const int n16  = lane & 15;      // column / row-in-tile index
    const int m0   = (wave >> 1) * 16;  // M tile rows [m0, m0+16)
    const int nh   = wave & 1;          // N half: tiles nh*8 .. nh*8+7
    const int b0   = blockIdx.x * BTILE;

    // ---- Prologue: W1 f32 -> f16 in LDS, staged via async copy to LDS ----
    {
        float* stage = (float*)(smem + SPK_OFF);   // 64KB staging in spike region
        for (int c = 0; c < 2; ++c) {              // 2 chunks of 16384 floats
            #pragma unroll
            for (int p = 0; p < 32; ++p) {
                uint32_t fe = (uint32_t)(p * NTHREADS + tid) * 4u;
                uint64_t ga = (uint64_t)(uintptr_t)(W1 + c * 16384 + fe);
                uint32_t la = (uint32_t)(uintptr_t)(stage + fe);
                asm volatile("global_load_async_to_lds_b128 %0, %1, off"
                             :: "v"(la), "v"(ga) : "memory");
            }
            asm volatile("s_wait_asynccnt 0x0" ::: "memory");
            __syncthreads();
            #pragma unroll 8
            for (int j = 0; j < 128; ++j) {
                int f  = j * NTHREADS + tid;
                int gi = c * 16384 + f;
                int hr = gi >> 7, k = gi & 127;
                w1l[hr * W1_STRIDE + k] = (_Float16)stage[f];
            }
            __syncthreads();
        }
    }

    // ---- Poisson encoding for ALL timesteps: each x element read once ----
    for (int jj = 0; jj < (BTILE * IN_DIM) / NTHREADS; ++jj) {
        int e = jj * NTHREADS + tid;
        int r = e >> 7, i = e & 127;
        float xv = x[(b0 + r) * IN_DIM + i];
        #pragma unroll
        for (int t = 0; t < T_STEPS; ++t) {
            uint32_t idx = ((uint32_t)t * B_TOT + (uint32_t)(b0 + r)) * IN_DIM + i;
            spk[t * SPK_TSTRIDE + r * SPK_STRIDE + i] =
                (urand(idx) < xv) ? (_Float16)1.0f : (_Float16)0.0f;
        }
    }
    __syncthreads();

    float b1v[8];
    #pragma unroll
    for (int nt = 0; nt < 8; ++nt)
        b1v[nt] = b1[(nh * 8 + nt) * 16 + n16];

    // A fragment base (16x32 f16, ISA 7.12.2): lanes 0-15 hold K=0..7,16..23;
    // lanes 16-31 hold K=8..15,24..31 — two 16B DS loads per K-step fragment.
    const _Float16* arow = spk + (m0 + n16) * SPK_STRIDE + 8 * g;

    v8f acc[8];                      // per-tile spike counts, register-resident
    #pragma unroll
    for (int nt = 0; nt < 8; ++nt) {
        const int hrow = (nh * 8 + nt) * 16 + n16;   // W1 row = output column

        // B fragments (32x16): lane = column, K = 8*g + 0..7, 16+8*g + 0..7. Loaded ONCE,
        // reused for all 10 timesteps.
        v16h bfr[4];
        #pragma unroll
        for (int kk = 0; kk < 4; ++kk) {
            const _Float16* base = w1l + hrow * W1_STRIDE + kk * 32 + 8 * g;
            v8h lo = *(const v8h*)(base);
            v8h hi = *(const v8h*)(base + 16);
            bfr[kk] = __builtin_shufflevector(lo, hi, 0,1,2,3,4,5,6,7,
                                                      8,9,10,11,12,13,14,15);
        }

        v8f v, a;
        #pragma unroll
        for (int r = 0; r < 8; ++r) { v[r] = 0.0f; a[r] = 0.0f; }

        v16h aA[4], aB[4];           // ping-pong A fragment buffers

        auto load_a = [&](v16h* dst, int t) {
            #pragma unroll
            for (int kk = 0; kk < 4; ++kk) {
                const _Float16* base = arow + t * SPK_TSTRIDE + kk * 32;
                v8h lo = *(const v8h*)(base);
                v8h hi = *(const v8h*)(base + 16);
                dst[kk] = __builtin_shufflevector(lo, hi, 0,1,2,3,4,5,6,7,
                                                          8,9,10,11,12,13,14,15);
            }
        };

        // GEMM tile + double LIF with hard reset; only 2nd spike accumulates.
        auto step = [&](const v16h* af) {
            v8f c;
            #pragma unroll
            for (int r = 0; r < 8; ++r) c[r] = b1v[nt];
            #pragma unroll
            for (int kk = 0; kk < 4; ++kk)
                c = __builtin_amdgcn_wmma_f32_16x16x32_f16(
                        false, af[kk], false, bfr[kk], (short)0, c,
                        false, false);
            #pragma unroll
            for (int r = 0; r < 8; ++r) {
                float vv = v[r];
                vv = BETA * vv + c[r];
                vv = (vv >= 1.0f) ? 0.0f : vv;          // lif1: reset, z unused
                vv = BETA * vv + c[r];
                float z = (vv >= 1.0f) ? 1.0f : 0.0f;   // lif2
                a[r] += z;
                v[r] = (z != 0.0f) ? 0.0f : vv;
            }
        };

        load_a(aA, 0);
        for (int t = 0; t < T_STEPS; t += 2) {   // T even: no copy rotation
            load_a(aB, t + 1);                   // prefetch t+1, compute t
            step(aA);
            int tpa = (t + 2 < T_STEPS) ? (t + 2) : (T_STEPS - 1);
            load_a(aA, tpa);                     // prefetch t+2, compute t+1
            step(aB);
        }
        acc[nt] = a;
    }

    // --------------------- epilogue: fc2 on summed spikes ------------------
    __syncthreads();                       // done reading w1l/spk; reuse LDS
    float* sacc = (float*)(smem);          // 32 x 260 f32 (overlaps W1 region)
    #pragma unroll
    for (int nt = 0; nt < 8; ++nt)
        #pragma unroll
        for (int r = 0; r < 8; ++r)
            sacc[(m0 + r + 8 * g) * ACC_STRIDE + (nh * 8 + nt) * 16 + n16] =
                acc[nt][r];
    __syncthreads();

    for (int idx = tid; idx < BTILE * OUT_DIM; idx += NTHREADS) {
        int row = idx / OUT_DIM;
        int o   = idx - row * OUT_DIM;
        const float* ar = sacc + row * ACC_STRIDE;
        const float* wr = W2 + o * H_DIM;
        float s = b2[o];
        #pragma unroll 8
        for (int h = 0; h < H_DIM; ++h) s += ar[h] * wr[h];
        out[(b0 + row) * OUT_DIM + o] = s;
    }
}

extern "C" void kernel_launch(void* const* d_in, const int* in_sizes, int n_in,
                              void* d_out, int out_size, void* d_ws, size_t ws_size,
                              hipStream_t stream) {
    (void)in_sizes; (void)n_in; (void)out_size; (void)d_ws; (void)ws_size;
    const float* x  = (const float*)d_in[0];
    const float* W1 = (const float*)d_in[1];
    const float* b1 = (const float*)d_in[2];
    const float* W2 = (const float*)d_in[3];
    const float* b2 = (const float*)d_in[4];
    float* out = (float*)d_out;
    snn_fused<<<B_TOT / BTILE, NTHREADS, 0, stream>>>(x, W1, b1, W2, b2, out);
}
